// yuyi_4578435137619
// MI455X (gfx1250) — matrix operand order repaired and checked by reading, not compiled, not run on hardware
//
#include <hip/hip_runtime.h>
#include <hip/hip_bf16.h>

typedef __attribute__((ext_vector_type(16))) __bf16 v16bf;
typedef __attribute__((ext_vector_type(8)))  float  v8f;
typedef unsigned short u16;

static constexpr int HWv   = 4096;   // 64*64
static constexpr int CINv  = 256;
static constexpr int COv   = 128;
static constexpr int Bv    = 4;

// ---------- helpers ----------
__device__ __forceinline__ u16 f2bf(float f) {
    unsigned u = __float_as_uint(f);
    u += 0x7fffu + ((u >> 16) & 1u);      // round-to-nearest-even
    return (u16)(u >> 16);
}
__device__ __forceinline__ float gelu_exact(float v) {
    return 0.5f * v * (1.0f + erff(v * 0.70710678118654752f));
}
__device__ __forceinline__ v8f v8f_zero() {
    v8f z;
#pragma unroll
    for (int i = 0; i < 8; ++i) z[i] = 0.0f;
    return z;
}
union FragU { unsigned u[8]; v16bf v; };

// A-matrix fragment (16xK tile row-major in memory of uint pairs, strideU uints per row)
// lane half h: K pairs at {k0+8h+2v} v=0..3 and {k0+16+8h+2v} v=0..3  (ISA 7.12.2, 16-bit A 16x32)
__device__ __forceinline__ v16bf frag_a(const unsigned* base, int row, int strideU, int koff2, int lane) {
    FragU f;
    const unsigned* p = base + (long)row * strideU + koff2 + ((lane >> 4) & 1) * 4;
#pragma unroll
    for (int i = 0; i < 4; ++i) f.u[i] = p[i];
#pragma unroll
    for (int i = 0; i < 4; ++i) f.u[4 + i] = p[8 + i];
    return f.v;
}
// B-matrix fragment from K-major rows (row = column n): lane half h covers K in [8h,8h+8) and [16+8h,16+8h+8)
__device__ __forceinline__ v16bf frag_b(const unsigned* base, int row, int strideU, int koff2, int lane) {
    FragU f;
    const unsigned* p = base + (long)row * strideU + koff2 + ((lane >> 4) & 1) * 4;
#pragma unroll
    for (int i = 0; i < 4; ++i) f.u[i] = p[i];
#pragma unroll
    for (int i = 0; i < 4; ++i) f.u[4 + i] = p[8 + i];
    return f.v;
}
__device__ __forceinline__ v8f wmma_bf16(v16bf a, v16bf b, v8f c) {
    return __builtin_amdgcn_wmma_f32_16x16x32_bf16(false, a, false, b, (short)0, c, false, false);
}

// ---- CDNA5 async global->LDS copy (ASYNCcnt path, CDNA5 ISA) ----
// GVS mode: mem_addr = SADDR(64) + VADDR(i32 byte offset); VDST = LDS byte address.
__device__ __forceinline__ void async_ld_b128(unsigned lds_addr, const void* gbase, unsigned byte_off) {
    asm volatile("global_load_async_to_lds_b128 %0, %1, %2"
                 :
                 : "v"(lds_addr), "v"(byte_off), "s"((unsigned long long)(uintptr_t)gbase)
                 : "memory");
}
__device__ __forceinline__ void wait_async0() {
    asm volatile("s_wait_asynccnt 0x0" ::: "memory");
}

// ---------- small kernels ----------
__global__ __launch_bounds__(256) void cvt_kernel(const float* __restrict__ s, u16* __restrict__ d, int n) {
    for (int i = blockIdx.x * 256 + threadIdx.x; i < n; i += gridDim.x * 256) d[i] = f2bf(s[i]);
}

__global__ __launch_bounds__(256) void stats_kernel(const float* __restrict__ x,
                                                    float* __restrict__ avg, float* __restrict__ mx) {
    int bc = blockIdx.x, t = threadIdx.x;
    const float* p = x + (long)bc * HWv;
    float s = 0.0f, m = -1e30f;
    for (int i = t; i < HWv; i += 256) { float v = p[i]; s += v; m = fmaxf(m, v); }
    __shared__ float ss[256], sm[256];
    ss[t] = s; sm[t] = m; __syncthreads();
    for (int d = 128; d; d >>= 1) {
        if (t < d) { ss[t] += ss[t + d]; sm[t] = fmaxf(sm[t], sm[t + d]); }
        __syncthreads();
    }
    if (t == 0) { avg[bc] = ss[0] * (1.0f / HWv); mx[bc] = sm[0]; }
}

// channel weighting MLP + pooled ASPP branch (tiny, one block)
__global__ __launch_bounds__(256) void cw_kernel(
    const float* __restrict__ avg, const float* __restrict__ mx,
    const float* __restrict__ w1, const float* __restrict__ b1,
    const float* __restrict__ lng, const float* __restrict__ lnb,
    const float* __restrict__ w2, const float* __restrict__ b2,
    const float* __restrict__ pw, const float* __restrict__ pg, const float* __restrict__ pb,
    float* __restrict__ sp1, float* __restrict__ bpv)
{
    __shared__ float o_s[256], h_s[128], gp_s[256], t_s[256], stat[2];
    int t = threadIdx.x;
    for (int b = 0; b < Bv; ++b) {
        float a = avg[b * 256 + t], m = mx[b * 256 + t];
        o_s[t] = fabsf(a - m) * a;
        __syncthreads();
        if (t < 128) {
            float acc = b1[t];
            for (int i = 0; i < 256; ++i) acc += o_s[i] * w1[t * 256 + i];
            h_s[t] = acc;
        }
        __syncthreads();
        if (t == 0) {
            float s = 0, s2 = 0;
            for (int i = 0; i < 128; ++i) { s += h_s[i]; s2 += h_s[i] * h_s[i]; }
            float mu = s / 128.0f;
            stat[0] = mu; stat[1] = rsqrtf(s2 / 128.0f - mu * mu + 1e-5f);
        }
        __syncthreads();
        if (t < 128) h_s[t] = lng[t] * (h_s[t] - stat[0]) * stat[1] + lnb[t];
        __syncthreads();
        {
            float acc = b2[t];
            for (int i = 0; i < 128; ++i) acc += h_s[i] * w2[t * 128 + i];
            float s = 1.0f / (1.0f + __expf(-acc));
            sp1[b * 256 + t] = 1.0f + s;
            gp_s[t] = a * (1.0f + s);          // pooled xc
        }
        __syncthreads();
        {
            float acc = 0.0f;
            for (int i = 0; i < 256; ++i) acc += gp_s[i] * pw[t * 256 + i];
            t_s[t] = acc;
        }
        __syncthreads();
        if (t == 0) {
            float s = 0, s2 = 0;
            for (int i = 0; i < 256; ++i) { s += t_s[i]; s2 += t_s[i] * t_s[i]; }
            float mu = s / 256.0f;
            stat[0] = mu; stat[1] = rsqrtf(s2 / 256.0f - mu * mu + 1e-6f);
        }
        __syncthreads();
        bpv[b * 256 + t] = gelu_exact(pg[t] * (t_s[t] - stat[0]) * stat[1] + pb[t]);
        __syncthreads();
    }
}

__global__ __launch_bounds__(256) void scale_kernel(const float* __restrict__ x, const float* __restrict__ sp1,
                                                    u16* __restrict__ x_bf, u16* __restrict__ xc_bf) {
    const long n = (long)Bv * CINv * HWv;
    for (long i = (long)blockIdx.x * 256 + threadIdx.x; i < n; i += (long)gridDim.x * 256) {
        float v = x[i];
        int b = (int)(i >> 20), c = (int)((i >> 12) & 255);
        x_bf[i]  = f2bf(v);
        xc_bf[i] = f2bf(v * sp1[b * 256 + c]);
    }
}

__global__ __launch_bounds__(256) void fill_bp_kernel(const float* __restrict__ bpv, u16* __restrict__ cat_bf) {
    const long n = (long)Bv * 256 * HWv;
    for (long i = (long)blockIdx.x * 256 + threadIdx.x; i < n; i += (long)gridDim.x * 256) {
        int b = (int)(i >> 20), co = (int)((i >> 12) & 255), sp = (int)(i & 4095);
        cat_bf[(long)b * 1280 * HWv + (long)(1024 + co) * HWv + sp] = f2bf(bpv[b * 256 + co]);
    }
}

// ---------- WMMA GEMM (1x1 / implicit dilated 3x3) with fused bias / channels-first LN + GELU ----------
// C[M, HW] = A[M, K] * Act[K, HW]  per batch; block computes all M rows for a 32-wide spatial tile.
// A tile is staged via GLOBAL_LOAD_ASYNC_TO_LDS_B128 (ASYNCcnt); B tile is a DS-path gather.
template <int MROWS>
__global__ __launch_bounds__(256) void gemm_kernel(
    const u16* __restrict__ A,            // [MROWS][K] bf16 weights
    const u16* __restrict__ act,          // [B][Cact][HW] bf16 activations
    long actBStride, int K, int dil,
    const float* __restrict__ bias,
    const float* __restrict__ ln_g, const float* __restrict__ ln_b,
    u16* __restrict__ out_bf, long outBStrideBf,
    float* __restrict__ out_f32, long outBStrideF)
{
    constexpr int MSUB = MROWS / 128;      // 16-row subtiles per wave (8 waves)
    __shared__ unsigned AsU[MROWS * 16];   // bf16 [MROWS][32]
    __shared__ unsigned BtU[32 * 16];      // bf16 [32 n][32 k]
    __shared__ float Cs[MROWS * 32];
    __shared__ float lnMean[32], lnRstd[32];

    const int t = threadIdx.x, lane = t & 31, wave = t >> 5, hl = (lane >> 4) & 1;
    const int n0 = blockIdx.x * 32, b = blockIdx.y;
    const u16* actb = act + (long)b * actBStride;
    const unsigned asBase = (unsigned)(uintptr_t)AsU;   // LDS byte address of A staging buffer

    v8f acc[MSUB][2];
#pragma unroll
    for (int ms = 0; ms < MSUB; ++ms) { acc[ms][0] = v8f_zero(); acc[ms][1] = v8f_zero(); }

    for (int k0 = 0; k0 < K; k0 += 32) {
        __syncthreads();
        // A tile: MROWS rows x 64B, async global->LDS in 16B chunks (MROWS*4 chunks)
        for (int i = t; i < MROWS * 4; i += 256) {
            int m = i >> 2, ch = i & 3;
            async_ld_b128(asBase + (unsigned)(m * 64 + ch * 16),
                          A, (unsigned)(((long)m * K + k0) * 2 + ch * 16));
        }
        u16* Bt16 = (u16*)BtU;
        if (dil == 0) {
            for (int i = t; i < 32 * 32; i += 256) {
                int k = i >> 5, n = i & 31;
                Bt16[n * 32 + k] = actb[(long)(k0 + k) * HWv + n0 + n];
            }
        } else {                                             // implicit dilated 3x3 gather
            for (int i = t; i < 32 * 32; i += 256) {
                int k = i >> 5, n = i & 31;
                int kk = k0 + k;
                int ci = kk / 9, rem = kk - ci * 9;
                int kh = rem / 3, kw = rem - kh * 3;
                int sp = n0 + n, h = sp >> 6, w = sp & 63;
                int sh = h + dil * (kh - 1), sw = w + dil * (kw - 1);
                u16 v = 0;
                if ((unsigned)sh < 64u && (unsigned)sw < 64u)
                    v = actb[(long)ci * HWv + (sh << 6) + sw];
                Bt16[n * 32 + k] = v;
            }
        }
        wait_async0();
        __syncthreads();
        v16bf bf0 = frag_b(BtU, 0 + (lane & 15), 16, 0, lane);
        v16bf bf1 = frag_b(BtU, 16 + (lane & 15), 16, 0, lane);
#pragma unroll
        for (int ms = 0; ms < MSUB; ++ms) {
            v16bf af = frag_a(AsU, wave * (16 * MSUB) + ms * 16 + (lane & 15), 16, 0, lane);
            acc[ms][0] = wmma_bf16(af, bf0, acc[ms][0]);
            acc[ms][1] = wmma_bf16(af, bf1, acc[ms][1]);
        }
    }
    __syncthreads();
#pragma unroll
    for (int ms = 0; ms < MSUB; ++ms)
#pragma unroll
        for (int ns = 0; ns < 2; ++ns)
#pragma unroll
            for (int r = 0; r < 8; ++r) {
                int m = wave * (16 * MSUB) + ms * 16 + r + hl * 8;
                int n = ns * 16 + (lane & 15);
                Cs[m * 32 + n] = acc[ms][ns][r];
            }
    __syncthreads();
    const bool doLN = (ln_g != nullptr);
    if (doLN && t < 32) {                                    // per-position stats over channels
        float s = 0, s2 = 0;
        for (int m = 0; m < MROWS; ++m) { float v = Cs[m * 32 + t]; s += v; s2 += v * v; }
        float mu = s / (float)MROWS;
        lnMean[t] = mu; lnRstd[t] = rsqrtf(s2 / (float)MROWS - mu * mu + 1e-6f);
    }
    __syncthreads();
    for (int i = t; i < MROWS * 32; i += 256) {
        int m = i >> 5, n = i & 31;
        float v = Cs[i];
        if (bias) v += bias[m];
        if (doLN) v = gelu_exact((v - lnMean[n]) * lnRstd[n] * ln_g[m] + ln_b[m]);
        long oidx = (long)m * HWv + n0 + n;
        if (out_bf)  out_bf[(long)b * outBStrideBf + oidx] = f2bf(v);
        if (out_f32) out_f32[(long)b * outBStrideF + oidx] = v;
    }
}

// ---------- attention pass 1: per-row online max / sumexp of S = Q^T P / 64 ----------
__global__ __launch_bounds__(256) void attn_rowstats(
    const u16* __restrict__ Qb, const u16* __restrict__ Pb,
    float* __restrict__ rowmax, float* __restrict__ rowsum)
{
    __shared__ unsigned QT[128 * 64];   // [q][c=128] bf16
    __shared__ unsigned PT[32 * 64];    // [p][c=128] bf16
    const int t = threadIdx.x, lane = t & 31, wave = t >> 5, hl = (lane >> 4) & 1;
    const int q0 = blockIdx.x * 128, b = blockIdx.y;
    const u16* Q = Qb + (long)b * COv * HWv;
    const u16* P = Pb + (long)b * COv * HWv;
    {
        u16* QT16 = (u16*)QT;
        for (int i = t; i < 128 * 128; i += 256) { int q = i >> 7, c = i & 127; QT16[q * 128 + c] = Q[(long)c * HWv + q0 + q]; }
    }
    float m8[8], s8[8];
#pragma unroll
    for (int r = 0; r < 8; ++r) { m8[r] = -1e30f; s8[r] = 0.0f; }
    const float rs = 0.015625f;  // 1/sqrt(4096)
    for (int p0 = 0; p0 < HWv; p0 += 32) {
        __syncthreads();
        u16* PT16 = (u16*)PT;
        for (int i = t; i < 32 * 128; i += 256) { int p = i >> 7, c = i & 127; PT16[p * 128 + c] = P[(long)c * HWv + p0 + p]; }
        __syncthreads();
        v8f a0 = v8f_zero(), a1 = v8f_zero();
#pragma unroll
        for (int ks = 0; ks < 4; ++ks) {
            v16bf af = frag_a(QT, wave * 16 + (lane & 15), 64, ks * 16, lane);
            v16bf b0 = frag_b(PT, 0 + (lane & 15), 64, ks * 16, lane);
            v16bf b1 = frag_b(PT, 16 + (lane & 15), 64, ks * 16, lane);
            a0 = wmma_bf16(af, b0, a0);
            a1 = wmma_bf16(af, b1, a1);
        }
#pragma unroll
        for (int r = 0; r < 8; ++r) {
            float v0 = a0[r] * rs, v1 = a1[r] * rs;
            float mn = fmaxf(m8[r], fmaxf(v0, v1));
            s8[r] = s8[r] * __expf(m8[r] - mn) + __expf(v0 - mn) + __expf(v1 - mn);
            m8[r] = mn;
        }
    }
#pragma unroll
    for (int r = 0; r < 8; ++r) {                // reduce across the 16 lanes of each half-wave
        float m = m8[r], s = s8[r];
#pragma unroll
        for (int d = 1; d < 16; d <<= 1) {
            float mo = __shfl_xor(m, d, 32);
            float so = __shfl_xor(s, d, 32);
            float mn = fmaxf(m, mo);
            s = s * __expf(m - mn) + so * __expf(mo - mn);
            m = mn;
        }
        if ((lane & 15) == 0) {
            int q = q0 + wave * 16 + r + hl * 8;
            rowmax[(long)b * HWv + q] = m;
            rowsum[(long)b * HWv + q] = s;
        }
    }
}

// ---------- attention pass 2: O = P * softmax(S) + xa  (flash-style, chained WMMAs) ----------
__global__ __launch_bounds__(256) void attn_apply(
    const u16* __restrict__ Qb, const u16* __restrict__ Pb,
    const float* __restrict__ rowmax, const float* __restrict__ rowsum,
    const float* __restrict__ xa, u16* __restrict__ Obf)
{
    __shared__ unsigned PTj[32 * 64];   // [j][c]
    __shared__ unsigned QT[64 * 64];    // [q][c]
    __shared__ unsigned Wt[32 * 32];    // [j][64 q] bf16
    __shared__ float mzs[128];          // m[64], z[64]
    const int t = threadIdx.x, lane = t & 31, wave = t >> 5, hl = (lane >> 4) & 1;
    const int j0 = blockIdx.x * 32, b = blockIdx.y;
    const u16* Q = Qb + (long)b * COv * HWv;
    const u16* P = Pb + (long)b * COv * HWv;
    {
        u16* PTj16 = (u16*)PTj;
        for (int i = t; i < 32 * 128; i += 256) { int j = i >> 7, c = i & 127; PTj16[j * 128 + c] = P[(long)c * HWv + j0 + j]; }
    }
    v8f L[2] = { v8f_zero(), v8f_zero() };
    const float rs = 0.015625f;
    for (int qb = 0; qb < HWv; qb += 64) {
        __syncthreads();
        u16* QT16 = (u16*)QT;
        for (int i = t; i < 64 * 128; i += 256) { int q = i >> 7, c = i & 127; QT16[q * 128 + c] = Q[(long)c * HWv + qb + q]; }
        if (t < 64) { mzs[t] = rowmax[(long)b * HWv + qb + t]; mzs[64 + t] = rowsum[(long)b * HWv + qb + t]; }
        __syncthreads();
        // S subtile per wave: 16q x 16j over 64q x 32j
        const int qsub = (wave >> 1) * 16, jsub = (wave & 1) * 16;
        v8f s = v8f_zero();
#pragma unroll
        for (int ks = 0; ks < 4; ++ks) {
            v16bf af = frag_a(QT, qsub + (lane & 15), 64, ks * 16, lane);
            v16bf bf = frag_b(PTj, jsub + (lane & 15), 64, ks * 16, lane);
            s = wmma_bf16(af, bf, s);
        }
        u16* Wt16 = (u16*)Wt;
#pragma unroll
        for (int r = 0; r < 8; ++r) {
            int q = qsub + r + hl * 8;
            int j = jsub + (lane & 15);
            float w = __expf(s[r] * rs - mzs[q]) / mzs[64 + q];
            Wt16[j * 64 + q] = f2bf(w);
        }
        __syncthreads();
        // L[c, j] += P[c, qb..qb+64] * W   (A straight from global, K contiguous)
        const unsigned* Pg = (const unsigned*)(P + qb);
        const int mb = wave * 16;
#pragma unroll
        for (int ks = 0; ks < 2; ++ks) {
            v16bf af = frag_a(Pg, mb + (lane & 15), HWv / 2, ks * 16, lane);
            v16bf b0 = frag_b(Wt, 0 + (lane & 15), 32, ks * 16, lane);
            v16bf b1 = frag_b(Wt, 16 + (lane & 15), 32, ks * 16, lane);
            L[0] = wmma_bf16(af, b0, L[0]);
            L[1] = wmma_bf16(af, b1, L[1]);
        }
    }
#pragma unroll
    for (int ns = 0; ns < 2; ++ns)
#pragma unroll
        for (int r = 0; r < 8; ++r) {
            int c = wave * 16 + r + hl * 8;
            int j = j0 + ns * 16 + (lane & 15);
            float o = L[ns][r] + xa[(long)b * COv * HWv + (long)c * HWv + j];
            Obf[(long)b * COv * HWv + (long)c * HWv + j] = f2bf(o);
        }
}

// ---------- host ----------
extern "C" void kernel_launch(void* const* d_in, const int* in_sizes, int n_in,
                              void* d_out, int out_size, void* d_ws, size_t ws_size,
                              hipStream_t stream) {
    const float* x       = (const float*)d_in[0];
    const float* conv1_w = (const float*)d_in[1];
    const float* conv1_b = (const float*)d_in[2];
    const float* conv2_w = (const float*)d_in[3];
    const float* conv2_b = (const float*)d_in[4];
    const float* cw_w1   = (const float*)d_in[5];
    const float* cw_b1   = (const float*)d_in[6];
    const float* cw_ln_g = (const float*)d_in[7];
    const float* cw_ln_b = (const float*)d_in[8];
    const float* cw_w2   = (const float*)d_in[9];
    const float* cw_b2   = (const float*)d_in[10];
    const float* aspp0_w = (const float*)d_in[11];
    const float* aspp0_g = (const float*)d_in[12];
    const float* aspp0_b = (const float*)d_in[13];
    const float* aspp1_w = (const float*)d_in[14];
    const float* aspp1_g = (const float*)d_in[15];
    const float* aspp1_b = (const float*)d_in[16];
    const float* aspp2_w = (const float*)d_in[17];
    const float* aspp2_g = (const float*)d_in[18];
    const float* aspp2_b = (const float*)d_in[19];
    const float* aspp3_w = (const float*)d_in[20];
    const float* aspp3_g = (const float*)d_in[21];
    const float* aspp3_b = (const float*)d_in[22];
    const float* asppp_w = (const float*)d_in[23];
    const float* asppp_g = (const float*)d_in[24];
    const float* asppp_b = (const float*)d_in[25];
    const float* proj_w  = (const float*)d_in[26];
    const float* proj_g  = (const float*)d_in[27];
    const float* proj_b  = (const float*)d_in[28];
    float* out = (float*)d_out;

    size_t off = 0;
    auto alloc = [&](size_t bytes) -> char* {
        char* p = (char*)d_ws + off;
        off += (bytes + 255) & ~(size_t)255;
        return p;
    };
    u16*   x_bf    = (u16*)alloc((size_t)Bv * CINv * HWv * 2);
    u16*   xc_bf   = (u16*)alloc((size_t)Bv * CINv * HWv * 2);
    u16*   cat_bf  = (u16*)alloc((size_t)Bv * 1280 * HWv * 2);
    u16*   proj_bf = (u16*)alloc((size_t)Bv * CINv * HWv * 2);
    u16*   P_bf    = (u16*)alloc((size_t)Bv * COv * HWv * 2);   // x1
    u16*   Q_bf    = (u16*)alloc((size_t)Bv * COv * HWv * 2);   // xa
    float* xa_f    = (float*)alloc((size_t)Bv * COv * HWv * 4);
    u16*   O_bf    = (u16*)alloc((size_t)Bv * COv * HWv * 2);
    u16*   wc1     = (u16*)alloc((size_t)COv * CINv * 2);
    u16*   wc2     = (u16*)alloc((size_t)CINv * COv * 2);
    u16*   wa0     = (u16*)alloc((size_t)CINv * CINv * 2);
    u16*   wa1     = (u16*)alloc((size_t)CINv * CINv * 9 * 2);
    u16*   wa2     = (u16*)alloc((size_t)CINv * CINv * 9 * 2);
    u16*   wa3     = (u16*)alloc((size_t)CINv * CINv * 9 * 2);
    u16*   wpr     = (u16*)alloc((size_t)CINv * 1280 * 2);
    float* avg_    = (float*)alloc((size_t)Bv * CINv * 4);
    float* mx_     = (float*)alloc((size_t)Bv * CINv * 4);
    float* sp1     = (float*)alloc((size_t)Bv * CINv * 4);
    float* bpv     = (float*)alloc((size_t)Bv * CINv * 4);
    float* rmax    = (float*)alloc((size_t)Bv * HWv * 4);
    float* rsum    = (float*)alloc((size_t)Bv * HWv * 4);
    (void)ws_size; (void)n_in; (void)in_sizes; (void)out_size;

    auto cvt = [&](const float* s, u16* d, int n) {
        cvt_kernel<<<(n + 255) / 256, 256, 0, stream>>>(s, d, n);
    };
    cvt(conv1_w, wc1, COv * CINv);
    cvt(conv2_w, wc2, CINv * COv);
    cvt(aspp0_w, wa0, CINv * CINv);
    cvt(aspp1_w, wa1, CINv * CINv * 9);
    cvt(aspp2_w, wa2, CINv * CINv * 9);
    cvt(aspp3_w, wa3, CINv * CINv * 9);
    cvt(proj_w,  wpr, CINv * 1280);

    stats_kernel<<<Bv * CINv, 256, 0, stream>>>(x, avg_, mx_);
    cw_kernel<<<1, 256, 0, stream>>>(avg_, mx_, cw_w1, cw_b1, cw_ln_g, cw_ln_b, cw_w2, cw_b2,
                                     asppp_w, asppp_g, asppp_b, sp1, bpv);
    scale_kernel<<<4096, 256, 0, stream>>>(x, sp1, x_bf, xc_bf);

    const dim3 ggrid(HWv / 32, Bv);
    const long actC  = (long)CINv * HWv;
    const long act5C = (long)1280 * HWv;
    const long actCo = (long)COv * HWv;

    // x1 = conv1(x) + b
    gemm_kernel<128><<<ggrid, 256, 0, stream>>>(wc1, x_bf, actC, CINv, 0, conv1_b,
                                                nullptr, nullptr, P_bf, actCo, nullptr, 0);
    // ASPP branches -> concat buffer (LN + GELU fused)
    gemm_kernel<256><<<ggrid, 256, 0, stream>>>(wa0, xc_bf, actC, CINv, 0, nullptr,
                                                aspp0_g, aspp0_b, cat_bf, act5C, nullptr, 0);
    gemm_kernel<256><<<ggrid, 256, 0, stream>>>(wa1, xc_bf, actC, CINv * 9, 3, nullptr,
                                                aspp1_g, aspp1_b, cat_bf + 1 * (size_t)CINv * HWv, act5C, nullptr, 0);
    gemm_kernel<256><<<ggrid, 256, 0, stream>>>(wa2, xc_bf, actC, CINv * 9, 6, nullptr,
                                                aspp2_g, aspp2_b, cat_bf + 2 * (size_t)CINv * HWv, act5C, nullptr, 0);
    gemm_kernel<256><<<ggrid, 256, 0, stream>>>(wa3, xc_bf, actC, CINv * 9, 9, nullptr,
                                                aspp3_g, aspp3_b, cat_bf + 3 * (size_t)CINv * HWv, act5C, nullptr, 0);
    fill_bp_kernel<<<4096, 256, 0, stream>>>(bpv, cat_bf);
    // proj (K=1280, LN + GELU)
    gemm_kernel<256><<<ggrid, 256, 0, stream>>>(wpr, cat_bf, act5C, 1280, 0, nullptr,
                                                proj_g, proj_b, proj_bf, actC, nullptr, 0);
    // xa = conv1(proj) + b  (bf16 for attention + f32 for residual)
    gemm_kernel<128><<<ggrid, 256, 0, stream>>>(wc1, proj_bf, actC, CINv, 0, conv1_b,
                                                nullptr, nullptr, Q_bf, actCo, xa_f, actCo);
    // attention (2-pass online softmax)
    attn_rowstats<<<dim3(HWv / 128, Bv), 256, 0, stream>>>(Q_bf, P_bf, rmax, rsum);
    attn_apply<<<dim3(HWv / 32, Bv), 256, 0, stream>>>(Q_bf, P_bf, rmax, rsum, xa_f, O_bf);
    // out = conv2(O) + b
    gemm_kernel<256><<<ggrid, 256, 0, stream>>>(wc2, O_bf, actCo, COv, 0, conv2_b,
                                                nullptr, nullptr, nullptr, 0, out, actC);
}
